// QuantLinear_52819507806398
// MI455X (gfx1250) — matrix operand order repaired and checked by reading, not compiled, not run on hardware
//
#include <hip/hip_runtime.h>

typedef __attribute__((ext_vector_type(16))) _Float16 v16h;
typedef __attribute__((ext_vector_type(8)))  _Float16 v8h;
typedef __attribute__((ext_vector_type(2)))  _Float16 v2h;
typedef __attribute__((ext_vector_type(8)))  float    v8f;

// GPTQ int4 fused dequant + GEMM for B=32.
//   out[32, N] = x[32, K] @ W[K, N] + bias,  W[k,n] = s[g,n] * (q[k,n] - z[g,n] - 1)
//
// One wave owns a 16-wide N tile and loops K in steps of 32, issuing two
// v_wmma_f32_16x16x32_f16 per step (M rows 0-15 and 16-31) that share one
// dequantized B fragment.
//
// Dequant trick (exact): fp16 bit pattern (0x6400 | q) == 1024 + q for
// q in [0,15]. So  W = s * (bitcast16(0x6400|q) - (1024 + z + 1)); the
// subtraction is exact on the fp16 integer grid, matching the reference's
// int-subtract-then-scale math; lowers to v_or3_b32 + v_pk_add_f16(neg) +
// v_pk_mul_f16(op_sel broadcast).
template <int STEPS_PER_GROUP>
__global__ __launch_bounds__(128)
void QuantLinear_52819507806398_kernel(const _Float16* __restrict__ x,
                                       const int*      __restrict__ qweight,  // [K/8, N]
                                       const int*      __restrict__ qzeros,   // [G, N/8]
                                       const _Float16* __restrict__ scales,   // [G, N]
                                       const _Float16* __restrict__ bias,     // [N]
                                       _Float16*       __restrict__ out,      // [32, N]
                                       int K, int N, int numGroups)
{
    const int lane   = threadIdx.x & 31;
    const int waveId = threadIdx.x >> 5;
    const int wavesPerBlock = blockDim.x >> 5;

    const int nTile = blockIdx.x * wavesPerBlock + waveId;   // 16-column tile
    if (nTile * 16 >= N) return;                             // wave-uniform exit

    const int l  = lane & 15;        // column in tile / row in A fragment
    const int hi = lane >> 4;        // half-wave select
    const int n  = nTile * 16 + l;   // global column (B/C fragments)
    const int kb = hi * 8;           // A-fragment K sub-base for this half-wave
    const int N8 = N >> 3;

    // Streaming pointers (constant-stride increments; no per-iter index mul).
    const _Float16* xp0 = x + (size_t)l        * K + kb;       // += 32 per K-step
    const _Float16* xp1 = x + (size_t)(l + 16) * K + kb;       // += 32 per K-step
    const int*      qwp = qweight + (size_t)hi * N + n;        // += 4N per K-step
    const int*      qzp = qzeros + (n >> 3);                   // += N/8 per group
    const _Float16* scp = scales + n;                          // += N per group
    const size_t qwRow  = (size_t)N;        // one packed row (8 K-values)
    const size_t qwStep = 4 * qwRow;        // 32 K-values

    v8f acc0 = {};   // rows 0..15
    v8f acc1 = {};   // rows 16..31

    for (int g = 0; g < numGroups; ++g) {
        // Per-group dequant constants for this lane's column.
        const _Float16 s   = *scp;
        const int      zq  = *qzp;
        const int      zp1 = ((zq >> ((n & 7) * 4)) & 0xF) + 1;
        const _Float16 off = (_Float16)(short)(1024 + zp1);   // exact in fp16
        v2h s2;   s2[0] = s;   s2[1] = s;
        v2h off2; off2[0] = off; off2[1] = off;

        #pragma unroll
        for (int step = 0; step < STEPS_PER_GROUP; ++step) {
            // ---- A fragments (16-bit A 16x32 layout): two b128 loads each ----
            const v8h a0lo = *(const v8h*)(xp0);
            const v8h a0hi = *(const v8h*)(xp0 + 16);
            const v8h a1lo = *(const v8h*)(xp1);
            const v8h a1hi = *(const v8h*)(xp1 + 16);
            v16h a0, a1;
            #pragma unroll
            for (int i = 0; i < 8; ++i) {
                a0[i] = a0lo[i]; a0[8 + i] = a0hi[i];
                a1[i] = a1lo[i]; a1[8 + i] = a1hi[i];
            }

            // ---- B fragment: 32 K-values of column n, packed-f16 dequant ----
            // Low half-wave: K 0-7,16-23 (rows r0, r0+2); high: K 8-15,24-31 (r0+1, r0+3).
            const unsigned qa = (unsigned)qwp[0];
            const unsigned qb = (unsigned)qwp[2 * qwRow];

            // Prefetch next K-step's rows. Condition is wave-uniform (scalar
            // counters only): compile-time true except on the group's last
            // step, where it's a scalar s_cmp — no per-lane v_cmpx/EXEC churn.
            if (step < STEPS_PER_GROUP - 1 || g < numGroups - 1)
                __builtin_prefetch(qwp + qwStep, 0, 1);

            const unsigned evA = qa & 0x0F0F0F0Fu;          // even nibbles -> bytes
            const unsigned odA = (qa >> 4) & 0x0F0F0F0Fu;   // odd  nibbles -> bytes
            const unsigned evB = qb & 0x0F0F0F0Fu;
            const unsigned odB = (qb >> 4) & 0x0F0F0F0Fu;

            v16h bfrag;
            #pragma unroll
            for (int j = 0; j < 4; ++j) {
                const unsigned pA = 0x64006400u
                                  | ((evA >> (8 * j)) & 0xFFu)
                                  | (((odA >> (8 * j)) & 0xFFu) << 16);
                const unsigned pB = 0x64006400u
                                  | ((evB >> (8 * j)) & 0xFFu)
                                  | (((odB >> (8 * j)) & 0xFFu) << 16);
                const v2h rA = (__builtin_bit_cast(v2h, pA) - off2) * s2;
                const v2h rB = (__builtin_bit_cast(v2h, pB) - off2) * s2;
                bfrag[2 * j]         = rA[0];
                bfrag[2 * j + 1]     = rA[1];
                bfrag[8 + 2 * j]     = rB[0];
                bfrag[8 + 2 * j + 1] = rB[1];
            }

            // ---- D = A x B + C : two WMMAs sharing bfrag ----
            acc0 = __builtin_amdgcn_wmma_f32_16x16x32_f16(
                false, a0, false, bfrag, (short)0, acc0, false, false);
            acc1 = __builtin_amdgcn_wmma_f32_16x16x32_f16(
                false, a1, false, bfrag, (short)0, acc1, false, false);

            xp0 += 32;
            xp1 += 32;
            qwp += qwStep;
        }
        qzp += N8;
        scp += N;
    }

    // ---- Epilogue: C layout lane=column n, VGPR j -> row j + 8*hi ----
    const _Float16 bv = bias[n];
    #pragma unroll
    for (int j = 0; j < 8; ++j) {
        const int m = j + 8 * hi;
        out[(size_t)m        * N + n] = (_Float16)acc0[j] + bv;  // rows 0..15
        out[(size_t)(m + 16) * N + n] = (_Float16)acc1[j] + bv;  // rows 16..31
    }
}

extern "C" void kernel_launch(void* const* d_in, const int* in_sizes, int n_in,
                              void* d_out, int out_size, void* d_ws, size_t ws_size,
                              hipStream_t stream) {
    (void)n_in; (void)out_size; (void)d_ws; (void)ws_size;
    const _Float16* x       = (const _Float16*)d_in[0];
    const int*      qweight = (const int*)d_in[1];
    const int*      qzeros  = (const int*)d_in[2];
    const _Float16* scales  = (const _Float16*)d_in[3];
    const _Float16* bias    = (const _Float16*)d_in[4];
    _Float16*       out     = (_Float16*)d_out;

    const int N  = in_sizes[4];               // 11008
    const int K  = (in_sizes[1] / N) * 8;     // 4096
    const int G  = in_sizes[3] / N;           // 32
    const int groupsize = K / G;              // 128
    const int steps = groupsize / 32;         // 4

    const int nTiles = N / 16;                // 688
    const int wavesPerBlock = 4;              // 128 threads
    dim3 block(32 * wavesPerBlock);
    dim3 grid((nTiles + wavesPerBlock - 1) / wavesPerBlock);

    switch (steps) {
    case 1:
        QuantLinear_52819507806398_kernel<1><<<grid, block, 0, stream>>>(
            x, qweight, qzeros, scales, bias, out, K, N, G);
        break;
    case 2:
        QuantLinear_52819507806398_kernel<2><<<grid, block, 0, stream>>>(
            x, qweight, qzeros, scales, bias, out, K, N, G);
        break;
    default:
        QuantLinear_52819507806398_kernel<4><<<grid, block, 0, stream>>>(
            x, qweight, qzeros, scales, bias, out, K, N, G);
        break;
    }
}
